// ConvTranBackbone_68298569941572
// MI455X (gfx1250) — matrix operand order repaired and checked by reading, not compiled, not run on hardware
//
#include <hip/hip_runtime.h>
#include <hip/hip_bf16.h>
#include <math.h>

#define B_   8
#define CIN_ 16
#define S_   1024
#define E_   128
#define H_   8
#define DH_  16
#define DFF_ 512
#define OC1_ 512
#define K2_  8192
#define EPS_BN 1e-5f
#define EPS_LN 1e-5f

typedef __attribute__((ext_vector_type(16))) _Float16 v16h;
typedef __attribute__((ext_vector_type(8)))  float    v8f;

// ---------------------------------------------------------------- helpers

__device__ __forceinline__ v8f wmma_f16(v16h a, v16h b, v8f c) {
  // v_wmma_f32_16x16x32_f16 : D = A(16x32) * B(32x16) + C(16x16 f32)
  return __builtin_amdgcn_wmma_f32_16x16x32_f16(false, a, false, b, (short)0, c,
                                                false, false);
}

// A fragment (16x32 f16), row-major source, row stride lda (elements).
// Lane L: m = L&15, g = L>>4; halves j -> K = g*16+j
__device__ __forceinline__ v16h load_A_frag(const _Float16* p, int lda) {
  const int lane = threadIdx.x & 31;
  const _Float16* row = p + (size_t)(lane & 15) * lda;
  const int g = lane >> 4;
  v16h a;
#pragma unroll
  for (int j = 0; j < 16; ++j) a[j] = row[g * 16 + j];
  return a;
}

// B fragment (32x16) where B[k][n] = W[n][k]; W row-major (rows = n, cols = k).
// Lane L: n = L&15, g = L>>4; halves j -> K = g*16+j  (16 contiguous elements of row n)
__device__ __forceinline__ v16h load_Bt_frag(const _Float16* p, int ldw) {
  const int lane = threadIdx.x & 31;
  const _Float16* row = p + (size_t)(lane & 15) * ldw + (lane >> 4) * 16;
  v16h b;
#pragma unroll
  for (int j = 0; j < 16; ++j) b[j] = row[j];
  return b;
}

__device__ __forceinline__ float gelu_f(float x) {
  return 0.5f * x * (1.0f + erff(x * 0.70710678118654752440f));
}

// ---------------------------------------------------------------- fp32 -> fp16

__global__ void cvt_f16(const float* __restrict__ in, _Float16* __restrict__ out,
                        int nElem) {
  int i = blockIdx.x * 256 + threadIdx.x;
  if (i < nElem) out[i] = (_Float16)in[i];
}

// ---------------------------------------------------------------- fused conv stack
// conv1(1x7)+BN1+GELU fused into conv2 GEMM (M=128 e, N=16 s, K=8192=(o,c)),
// conv1 activations never touch HBM. h1 K-chunk kept TRANSPOSED in LDS so the
// WMMA B fragment is 16 contiguous halfs per lane (2x ds_load_b128).
// Epilogue: BN2+GELU -> x_src, +tAPE -> x_pos f16.

__global__ void __launch_bounds__(256)
fused_conv_kernel(const float* __restrict__ x,
                  const float* __restrict__ c1w, const float* __restrict__ c1b,
                  const float* __restrict__ bn1g, const float* __restrict__ bn1b,
                  const float* __restrict__ bn1m, const float* __restrict__ bn1v,
                  const _Float16* __restrict__ w2h, const float* __restrict__ c2b,
                  const float* __restrict__ bn2g, const float* __restrict__ bn2b,
                  const float* __restrict__ bn2m, const float* __restrict__ bn2v,
                  float* __restrict__ x_src, _Float16* __restrict__ xph) {
  __shared__ float xs[CIN_][24];        // input window  s0-3 .. s0+18
  __shared__ float w1s[OC1_][8];        // conv1 taps
  __shared__ float sc1[OC1_], of1[OC1_];
  __shared__ _Float16 h1sT[16][32];     // transposed chunk: [n = s][k = (o-o0)*16+c]

  const int tid = threadIdx.x, wid = tid >> 5, lane = tid & 31;
  const int s0 = blockIdx.x * 16, b = blockIdx.y;

  for (int o = tid; o < OC1_; o += 256) {
    float s = bn1g[o] * rsqrtf(bn1v[o] + EPS_BN);
    sc1[o] = s;
    of1[o] = (c1b[o] - bn1m[o]) * s + bn1b[o];
#pragma unroll
    for (int t = 0; t < 7; ++t) w1s[o][t] = c1w[o * 7 + t];
  }
  for (int idx = tid; idx < CIN_ * 22; idx += 256) {
    int c = idx / 22, j = idx % 22;
    int sg = s0 - 3 + j;
    xs[c][j] = (sg >= 0 && sg < S_) ? x[((size_t)b * CIN_ + c) * S_ + sg] : 0.0f;
  }
  __syncthreads();

  v8f acc = {};
  for (int oc = 0; oc < 256; ++oc) {
    const int o0 = oc * 2;  // K chunk = 2 output channels x 16 c_in = 32
#pragma unroll
    for (int half = 0; half < 2; ++half) {
      int idx = tid + half * 256;
      int kl = idx & 31, n = idx >> 5;   // n: 0..15, kl: 0..31
      int o = o0 + (kl >> 4), c = kl & 15;
      float conv = 0.0f;
#pragma unroll
      for (int t = 0; t < 7; ++t) conv += xs[c][n + t] * w1s[o][t];
      h1sT[n][kl] = (_Float16)gelu_f(conv * sc1[o] + of1[o]);
    }
    __syncthreads();
    v16h a = load_A_frag(w2h + (size_t)(wid * 16) * K2_ + o0 * 16, K2_);
    v16h bf = load_Bt_frag(&h1sT[0][0], 32);  // contiguous per-lane LDS read
    acc = wmma_f16(a, bf, acc);
    __syncthreads();
  }

  const int n = lane & 15, g = lane >> 4;
#pragma unroll
  for (int r = 0; r < 8; ++r) {
    int e = wid * 16 + g * 8 + r;
    int s = s0 + n;
    float t = acc[r] + c2b[e];
    float s2 = bn2g[e] * rsqrtf(bn2v[e] + EPS_BN);
    t = (t - bn2m[e]) * s2 + bn2b[e];
    float val = gelu_f(t);
    size_t row = (size_t)b * S_ + s;
    x_src[row * E_ + e] = val;
    // tAPE: div = exp(-(ln1e4/E)*(2i)), ang = s*div*(E/S)
    float div = expf(-0.0719558153f * (float)(e & ~1));
    float ang = (float)s * div * ((float)E_ / (float)S_);
    float pe = (e & 1) ? cosf(ang) : sinf(ang);
    xph[row * E_ + e] = (_Float16)(val + pe);
  }
}

// ---------------------------------------------------------------- QKV projections
// out[b,s,e] = sum_k x_pos[b,s,k] * W[e,k]; M = B*S, N = 128, K = 128

__global__ void __launch_bounds__(256)
qkv_kernel(const _Float16* __restrict__ xp, const _Float16* __restrict__ wq,
           const _Float16* __restrict__ wk, const _Float16* __restrict__ wv,
           _Float16* __restrict__ q, _Float16* __restrict__ k,
           _Float16* __restrict__ v) {
  const int tid = threadIdx.x, wid = tid >> 5, lane = tid & 31;
  const int n = lane & 15, g = lane >> 4;
  const int mt = blockIdx.x, z = blockIdx.y;
  const _Float16* w = (z == 0) ? wq : (z == 1) ? wk : wv;
  _Float16* o = (z == 0) ? q : (z == 1) ? k : v;
  v8f acc = {};
#pragma unroll
  for (int ks = 0; ks < 4; ++ks) {
    v16h a = load_A_frag(xp + (size_t)(mt * 16) * E_ + ks * 32, E_);
    v16h bf = load_Bt_frag(w + (size_t)(wid * 16) * E_ + ks * 32, E_);
    acc = wmma_f16(a, bf, acc);
  }
#pragma unroll
  for (int r = 0; r < 8; ++r)
    o[(size_t)(mt * 16 + g * 8 + r) * E_ + wid * 16 + n] = (_Float16)acc[r];
}

// ---------------------------------------------------------------- attention
// One block per (b, h, 16-row i tile): scores strip (16x1024) via WMMA (K=16
// zero-padded to 32), softmax in f32, rel-bias added post-softmax, then
// (attn)x(V) via WMMA with K split across the 8 waves. V panel is staged
// transposed in LDS so phase-3 B fragments are contiguous LDS reads.

#define ATTN_SMEM (65536 + 32768 + 1024 + 128 + 8192 + 32768)

__global__ void __launch_bounds__(256)
attn_kernel(const _Float16* __restrict__ q_h, const _Float16* __restrict__ k_h,
            const _Float16* __restrict__ v_h, const float* __restrict__ relb,
            float* __restrict__ attn_out) {
  extern __shared__ char smem[];
  float* sc = (float*)smem;                       // [16][1024] f32 scores
  _Float16* ah = (_Float16*)(smem + 65536);       // [16][1024] f16 attn
  float* red = (float*)(smem + 65536 + 32768);    // [16][16]
  float* rst = red + 256;                         // rowmax[16], rowsum[16]
  float* part = rst + 32;                         // [8][256] partial C
  _Float16* vT = (_Float16*)(smem + 65536 + 32768 + 1024 + 128 + 8192); // [16][1024]

  const int tid = threadIdx.x, wid = tid >> 5, lane = tid & 31;
  const int n = lane & 15, g = lane >> 4;
  const int i0 = blockIdx.x * 16, h = blockIdx.y, bb = blockIdx.z;
  const float scale = 0.088388347648318447f;  // 1/sqrt(E)

  // phase 0: stage V panel (1024 x 16) transposed into LDS: vT[d][j]
  for (int j = tid; j < S_; j += 256) {
    const _Float16* vr = v_h + ((size_t)(bb * S_) + j) * E_ + h * 16;
    v16h tmp = *(const v16h*)vr;  // 32B-aligned vector load
#pragma unroll
    for (int d = 0; d < 16; ++d) vT[d * 1024 + j] = tmp[d];
  }

  // A fragment: q rows i0..i0+15, head h, K = DH=16 padded to 32 with zeros
  v16h qa;
  {
    const _Float16* qrow = q_h + ((size_t)(bb * S_) + i0 + n) * E_ + h * 16;
#pragma unroll
    for (int j = 0; j < 16; ++j) qa[j] = (g == 0) ? qrow[j] : (_Float16)0.0f;
  }

  // phase 1: score strip
  for (int jt = wid; jt < 64; jt += 8) {
    v16h kb;
    if (g == 0) {
      const _Float16* krow = k_h + ((size_t)(bb * S_) + jt * 16 + n) * E_ + h * 16;
#pragma unroll
      for (int j = 0; j < 16; ++j) kb[j] = krow[j];
    } else {
#pragma unroll
      for (int j = 0; j < 16; ++j) kb[j] = (_Float16)0.0f;
    }
    v8f c = {};
    c = wmma_f16(qa, kb, c);
#pragma unroll
    for (int r = 0; r < 8; ++r)
      sc[(g * 8 + r) * 1024 + jt * 16 + n] = c[r] * scale;
  }
  __syncthreads();

  // phase 2: softmax (+ rel-bias after softmax), store f16
  {
    const int r = tid >> 4, c = tid & 15;
    float mx = -3.4e38f;
    for (int j = c * 64; j < c * 64 + 64; ++j) mx = fmaxf(mx, sc[r * 1024 + j]);
    red[r * 16 + c] = mx;
    __syncthreads();
    if (c == 0) {
      float m = red[r * 16];
      for (int t = 1; t < 16; ++t) m = fmaxf(m, red[r * 16 + t]);
      rst[r] = m;
    }
    __syncthreads();
    float rm = rst[r], sm = 0.0f;
    for (int j = c * 64; j < c * 64 + 64; ++j) {
      float e = expf(sc[r * 1024 + j] - rm);
      sc[r * 1024 + j] = e;
      sm += e;
    }
    red[r * 16 + c] = sm;
    __syncthreads();
    if (c == 0) {
      float s = 0.0f;
      for (int t = 0; t < 16; ++t) s += red[r * 16 + t];
      rst[16 + r] = s;
    }
    __syncthreads();
    for (int idx = tid; idx < 16 * 1024; idx += 256) {
      int rr = idx >> 10, j = idx & 1023;
      float val = sc[idx] / rst[16 + rr] +
                  relb[(size_t)(i0 + rr - j + (S_ - 1)) * H_ + h];
      ah[idx] = (_Float16)val;
    }
    __syncthreads();
  }

  // phase 3: out = attn x V ; each wave owns 4 K-chunks of 32, reduce via LDS
  v8f acc = {};
  for (int kc = wid * 4; kc < wid * 4 + 4; ++kc) {
    v16h a = load_A_frag(ah + kc * 32, 1024);
    v16h bf = load_Bt_frag(vT + kc * 32, 1024);  // contiguous per-lane LDS read
    acc = wmma_f16(a, bf, acc);
  }
#pragma unroll
  for (int r = 0; r < 8; ++r) part[wid * 256 + lane * 8 + r] = acc[r];
  __syncthreads();
  {
    float v = 0.0f;
#pragma unroll
    for (int w = 0; w < 8; ++w) v += part[w * 256 + tid];
    int l2 = tid >> 3, r2 = tid & 7;
    int m = (l2 >> 4) * 8 + r2, nn = l2 & 15;
    attn_out[((size_t)(bb * S_) + i0 + m) * E_ + h * 16 + nn] = v;
  }
}

// ---------------------------------------------------------------- LN(attn) + residual + LN1

__global__ void __launch_bounds__(256)
ln_kernel(const float* __restrict__ aout, const float* __restrict__ xsrc,
          const float* __restrict__ g0, const float* __restrict__ b0,
          const float* __restrict__ g1, const float* __restrict__ b1,
          float* __restrict__ att, _Float16* __restrict__ atth) {
  const int tid = threadIdx.x, wid = tid >> 5, lane = tid & 31;
  const size_t row = (size_t)blockIdx.x * 8 + wid;
  const float* pr = aout + row * E_;
  float v[4];
#pragma unroll
  for (int i = 0; i < 4; ++i) v[i] = pr[lane * 4 + i];
  float s = 0.0f, sq = 0.0f;
#pragma unroll
  for (int i = 0; i < 4; ++i) { s += v[i]; sq += v[i] * v[i]; }
#pragma unroll
  for (int o = 16; o > 0; o >>= 1) {
    s += __shfl_xor(s, o);
    sq += __shfl_xor(sq, o);
  }
  float mean = s * (1.0f / E_);
  float inv = rsqrtf(sq * (1.0f / E_) - mean * mean + EPS_LN);
  const float* xr = xsrc + row * E_;
  float t[4];
#pragma unroll
  for (int i = 0; i < 4; ++i) {
    int e = lane * 4 + i;
    t[i] = xr[e] + (v[i] - mean) * inv * g0[e] + b0[e];
  }
  s = 0.0f; sq = 0.0f;
#pragma unroll
  for (int i = 0; i < 4; ++i) { s += t[i]; sq += t[i] * t[i]; }
#pragma unroll
  for (int o = 16; o > 0; o >>= 1) {
    s += __shfl_xor(s, o);
    sq += __shfl_xor(sq, o);
  }
  float mean2 = s * (1.0f / E_);
  float inv2 = rsqrtf(sq * (1.0f / E_) - mean2 * mean2 + EPS_LN);
#pragma unroll
  for (int i = 0; i < 4; ++i) {
    int e = lane * 4 + i;
    float r = (t[i] - mean2) * inv2 * g1[e] + b1[e];
    att[row * E_ + e] = r;
    atth[row * E_ + e] = (_Float16)r;
  }
}

// ---------------------------------------------------------------- FFN layer 1
// relu(att @ ff_w1^T + b1) ; M = B*S, N = 512, K = 128

__global__ void __launch_bounds__(256)
ff1_kernel(const _Float16* __restrict__ atth, const _Float16* __restrict__ w1h,
           const float* __restrict__ b1, _Float16* __restrict__ ff1h) {
  const int tid = threadIdx.x, wid = tid >> 5, lane = tid & 31;
  const int n = lane & 15, g = lane >> 4;
  const int mt = blockIdx.x, nt = blockIdx.y * 8 + wid;
  v8f acc = {};
#pragma unroll
  for (int ks = 0; ks < 4; ++ks) {
    v16h a = load_A_frag(atth + (size_t)(mt * 16) * E_ + ks * 32, E_);
    v16h bf = load_Bt_frag(w1h + (size_t)(nt * 16) * E_ + ks * 32, E_);
    acc = wmma_f16(a, bf, acc);
  }
#pragma unroll
  for (int r = 0; r < 8; ++r) {
    int col = nt * 16 + n;
    float vv = fmaxf(acc[r] + b1[col], 0.0f);
    ff1h[(size_t)(mt * 16 + g * 8 + r) * DFF_ + col] = (_Float16)vv;
  }
}

// ---------------------------------------------------------------- FFN layer 2 + residual + LN2 + transpose
// M = B*S (32 rows/block), N = 128, K = 512; epilogue LN over N, y written (B,E,S)

__global__ void __launch_bounds__(256)
ff2_kernel(const _Float16* __restrict__ ff1h, const _Float16* __restrict__ w2h,
           const float* __restrict__ b2, const float* __restrict__ att,
           const float* __restrict__ g2, const float* __restrict__ bb2,
           float* __restrict__ out) {
  __shared__ float buf[32][128];
  __shared__ float ps[32][8], pq[32][8], rm[32], rv[32];
  const int tid = threadIdx.x, wid = tid >> 5, lane = tid & 31;
  const int n = lane & 15, g = lane >> 4;
  const int m0 = blockIdx.x * 32;
  const int nt = wid;
  v8f acc0 = {}, acc1 = {};
  for (int ks = 0; ks < 16; ++ks) {
    __builtin_prefetch(ff1h + (size_t)m0 * DFF_ + (ks + 1) * 32, 0, 0);
    v16h a0 = load_A_frag(ff1h + (size_t)m0 * DFF_ + ks * 32, DFF_);
    v16h a1 = load_A_frag(ff1h + (size_t)(m0 + 16) * DFF_ + ks * 32, DFF_);
    v16h bf = load_Bt_frag(w2h + (size_t)(nt * 16) * DFF_ + ks * 32, DFF_);
    acc0 = wmma_f16(a0, bf, acc0);
    acc1 = wmma_f16(a1, bf, acc1);
  }
#pragma unroll
  for (int r = 0; r < 8; ++r) {
    int m = g * 8 + r, e = nt * 16 + n;
    buf[m][e] = acc0[r] + b2[e] + att[(size_t)(m0 + m) * E_ + e];
    buf[16 + m][e] = acc1[r] + b2[e] + att[(size_t)(m0 + 16 + m) * E_ + e];
  }
  __syncthreads();
  {
    const int row = tid >> 3, c = tid & 7;
    float s = 0.0f, sq = 0.0f;
    for (int e = c * 16; e < c * 16 + 16; ++e) {
      float v = buf[row][e];
      s += v;
      sq += v * v;
    }
    ps[row][c] = s;
    pq[row][c] = sq;
    __syncthreads();
    if (c == 0) {
      float S0 = 0.0f, Q = 0.0f;
      for (int t = 0; t < 8; ++t) { S0 += ps[row][t]; Q += pq[row][t]; }
      float mean = S0 * (1.0f / E_);
      rm[row] = mean;
      rv[row] = Q * (1.0f / E_) - mean * mean;
    }
    __syncthreads();
  }
  for (int idx = tid; idx < 32 * 128; idx += 256) {
    int rr = idx >> 7, e = idx & 127;
    int R = m0 + rr, bI = R >> 10, sI = R & 1023;
    float val = (buf[rr][e] - rm[rr]) * rsqrtf(rv[rr] + EPS_LN) * g2[e] + bb2[e];
    out[((size_t)(bI * E_) + e) * S_ + sI] = val;
  }
}

// ---------------------------------------------------------------- host launcher

extern "C" void kernel_launch(void* const* d_in, const int* in_sizes, int n_in,
                              void* d_out, int out_size, void* d_ws,
                              size_t ws_size, hipStream_t stream) {
  (void)in_sizes; (void)n_in; (void)out_size; (void)ws_size;
  const float* x    = (const float*)d_in[0];
  const float* c1w  = (const float*)d_in[1];
  const float* c1b  = (const float*)d_in[2];
  const float* bn1g = (const float*)d_in[3];
  const float* bn1b = (const float*)d_in[4];
  const float* bn1m = (const float*)d_in[5];
  const float* bn1v = (const float*)d_in[6];
  const float* c2w  = (const float*)d_in[7];
  const float* c2b  = (const float*)d_in[8];
  const float* bn2g = (const float*)d_in[9];
  const float* bn2b = (const float*)d_in[10];
  const float* bn2m = (const float*)d_in[11];
  const float* bn2v = (const float*)d_in[12];
  const float* wq   = (const float*)d_in[13];
  const float* wk   = (const float*)d_in[14];
  const float* wv   = (const float*)d_in[15];
  const float* relb = (const float*)d_in[16];
  const float* lnag = (const float*)d_in[17];
  const float* lnab = (const float*)d_in[18];
  const float* ln1g = (const float*)d_in[19];
  const float* ln1b = (const float*)d_in[20];
  const float* ln2g = (const float*)d_in[21];
  const float* ln2b = (const float*)d_in[22];
  const float* ffw1 = (const float*)d_in[23];
  const float* ffb1 = (const float*)d_in[24];
  const float* ffw2 = (const float*)d_in[25];
  const float* ffb2 = (const float*)d_in[26];

  char* ws = (char*)d_ws;
  size_t off = 0;
  auto alloc = [&](size_t bytes) -> void* {
    void* p = ws + off;
    off = (off + bytes + 255) & ~(size_t)255;
    return p;
  };
  const size_t NSE = (size_t)B_ * S_ * E_;   // 1M elements
  float*    x_src = (float*)alloc(NSE * 4);
  _Float16* xph   = (_Float16*)alloc(NSE * 2);
  _Float16* qh    = (_Float16*)alloc(NSE * 2);
  _Float16* kh    = (_Float16*)alloc(NSE * 2);
  _Float16* vh    = (_Float16*)alloc(NSE * 2);
  float*    aoutb = (float*)alloc(NSE * 4);
  float*    attb  = (float*)alloc(NSE * 4);
  _Float16* atth  = (_Float16*)alloc(NSE * 2);
  _Float16* ff1h  = (_Float16*)alloc((size_t)B_ * S_ * DFF_ * 2);
  _Float16* w2h   = (_Float16*)alloc((size_t)E_ * K2_ * 2);
  _Float16* wqh   = (_Float16*)alloc((size_t)E_ * E_ * 2);
  _Float16* wkh   = (_Float16*)alloc((size_t)E_ * E_ * 2);
  _Float16* wvh   = (_Float16*)alloc((size_t)E_ * E_ * 2);
  _Float16* w1h   = (_Float16*)alloc((size_t)DFF_ * E_ * 2);
  _Float16* w2fh  = (_Float16*)alloc((size_t)E_ * DFF_ * 2);

  cvt_f16<<<dim3((E_ * K2_ + 255) / 256), 256, 0, stream>>>(c2w, w2h, E_ * K2_);
  cvt_f16<<<dim3((E_ * E_ + 255) / 256), 256, 0, stream>>>(wq, wqh, E_ * E_);
  cvt_f16<<<dim3((E_ * E_ + 255) / 256), 256, 0, stream>>>(wk, wkh, E_ * E_);
  cvt_f16<<<dim3((E_ * E_ + 255) / 256), 256, 0, stream>>>(wv, wvh, E_ * E_);
  cvt_f16<<<dim3((DFF_ * E_ + 255) / 256), 256, 0, stream>>>(ffw1, w1h, DFF_ * E_);
  cvt_f16<<<dim3((E_ * DFF_ + 255) / 256), 256, 0, stream>>>(ffw2, w2fh, E_ * DFF_);

  fused_conv_kernel<<<dim3(S_ / 16, B_), 256, 0, stream>>>(
      x, c1w, c1b, bn1g, bn1b, bn1m, bn1v, w2h, c2b, bn2g, bn2b, bn2m, bn2v,
      x_src, xph);

  qkv_kernel<<<dim3(B_ * S_ / 16, 3), 256, 0, stream>>>(xph, wqh, wkh, wvh, qh,
                                                        kh, vh);

  attn_kernel<<<dim3(S_ / 16, H_, B_), 256, ATTN_SMEM, stream>>>(qh, kh, vh,
                                                                 relb, aoutb);

  ln_kernel<<<dim3(B_ * S_ / 8), 256, 0, stream>>>(aoutb, x_src, lnag, lnab,
                                                   ln1g, ln1b, attb, atth);

  ff1_kernel<<<dim3(B_ * S_ / 16, 4), 256, 0, stream>>>(atth, w1h, ffb1, ff1h);

  ff2_kernel<<<dim3(B_ * S_ / 32), 256, 0, stream>>>(ff1h, w2fh, ffb2, attb,
                                                     ln2g, ln2b, (float*)d_out);
}
